// NoRollCaTentLayer_2594160247026
// MI455X (gfx1250) — matrix operand order repaired and checked by reading, not compiled, not run on hardware
//
#include <hip/hip_runtime.h>
#include <hip/hip_bf16.h>
#include <stdint.h>

// ---------------- problem constants ----------------
constexpr int T       = 4096;
constexpr int C       = 1024;
constexpr int FW      = 8192;
constexpr int NUM_INH = 256;
constexpr float EPS   = 1e-08f;

// ---------------- WMMA tiling constants ----------------
constexpr int NCH   = FW / 16;            // 512 16-tap chunks
constexpr int Q0    = -15;                // first chunk base (column 0 edge)
constexpr int NITER = 271;                // chunks [-15-j, 526-j] ⊇ [0,512) per column
// padded transposed x: true index range touched is [-4335, 8191]; left pad 4335 makes the
// per-iteration window base 16*(m0+Q)+240 a multiple of 32 halves (64B aligned).
constexpr int XPAD    = 4335;
constexpr int XSTRIDE = 12544;            // per-channel halves, mult of 256
constexpr size_t CX   = (size_t)C * XSTRIDE;   // halves per x copy (2 copies, shift 0/1)
// padded transposed wf: 32 zero chunks left, 16 right -> chunk index in [-32, 528) is
// always in-bounds, so B loads need no guard.
constexpr int WCHPAD   = 32;                       // left pad, chunks
constexpr int WSTRIDE  = (WCHPAD + NCH + 16) * 16; // 8960 halves per channel

typedef __attribute__((ext_vector_type(16))) _Float16 v16h;
typedef __attribute__((ext_vector_type(8)))  float    v8f;

union pack32 {                            // 32-byte fragment staging
    uint32_t u[8];
    uint4    q[2];
    v16h     h;
};

// ---------------- pass 1: per-channel filter norm + relu + f16, padded transpose ------
// wf_t[c][WCHPAD*16 + k] = (f16) max(w[k][c] / max(||w[:,c]||2, EPS), 0), zeros in pads
__global__ __launch_bounds__(256) void k_norm_wf(const float* __restrict__ w,
                                                 _Float16* __restrict__ wf_t) {
    const int c   = blockIdx.x;
    const int tid = threadIdx.x;
    __shared__ float red[256];
    float s = 0.f;
    for (int k = tid; k < FW; k += 256) {
        float v = w[(size_t)k * C + c];
        s += v * v;
    }
    red[tid] = s;
    __syncthreads();
    for (int off = 128; off > 0; off >>= 1) {
        if (tid < off) red[tid] += red[tid + off];
        __syncthreads();
    }
    float norm = sqrtf(red[0]);
    norm = norm < EPS ? EPS : norm;
    const float rn = 1.0f / norm;
    for (int kk = tid; kk < WSTRIDE; kk += 256) {    // 8960 = 35*256, exact
        const int k = kk - WCHPAD * 16;
        float v = 0.f;
        if (k >= 0 && k < FW) {
            v = w[(size_t)k * C + c] * rn;
            v = v > 0.f ? v : 0.f;
        }
        wf_t[(size_t)c * WSTRIDE + kk] = (_Float16)v;
    }
}

// ---------------- pass 2: x -> two parity-shifted, zero-padded f16 transposes --------
// copy p (p=0,1): x2[p*CX + c*XSTRIDE + s] = x[s - XPAD + p, c]  (0 outside [0,T))
__global__ __launch_bounds__(256) void k_x2(const float* __restrict__ x,
                                            _Float16* __restrict__ x2) {
    const size_t idx = (size_t)blockIdx.x * 256 + threadIdx.x;
    if (idx >= 2 * CX) return;
    const int p = (int)(idx / CX);
    const size_t r = idx - (size_t)p * CX;
    const int c = (int)(r / XSTRIDE);
    const int s = (int)(r % XSTRIDE);
    const int t = s - XPAD + p;
    float v = (t >= 0 && t < T) ? x[(size_t)t * C + c] : 0.f;
    x2[idx] = (_Float16)v;
}

// ---------------- pass 3: depthwise conv via shifted-filter Toeplitz WMMA -------------
// One wave: channel c, output rows [256g, 256g+256).  D[i,j] = out[16*(16g+j)+i].
// Iteration t (Q = Q0+2t):  A[i,r] = x[16(m0+Q)-4095 + i + r]   (shared 47-half window)
//                           B[r,j] = wf[16(Q-j) + r]            (chunks Q-j, Q-j+1 per column)
__global__ __launch_bounds__(256) void k_conv(const _Float16* __restrict__ x2,
                                              const _Float16* __restrict__ wf_t,
                                              const float* __restrict__ bias,
                                              float* __restrict__ out) {
    const int lane = threadIdx.x & 31;
    const int wid  = (blockIdx.x << 3) + (threadIdx.x >> 5);  // 8 waves / block
    const int c    = wid >> 4;                                // channel
    const int g    = wid & 15;                                // output tile-group
    const int j    = lane & 15;                               // N col (== A row M)
    const int h    = lane >> 4;                               // lane half

    // A lane layout (16-bit A 16x32): element e holds K = 8h+e+(e>=8?8:8*(e>=8)),
    // i.e. two runs of 8 halves at window offsets o1 = M+8h and o1+16.
    // Parity copy p makes the base even -> plain dword loads.
    const int o1  = j + (h << 3);
    const int p   = o1 & 1;
    const int o1e = o1 & ~1;

    // per-lane word (uint32) bases; both advance 16 words (64B) per iteration
    const uint32_t* __restrict__ xw =
        (const uint32_t*)x2 +
        (((size_t)p * CX + (size_t)c * XSTRIDE + ((size_t)g << 8) + (size_t)o1e) >> 1);
    const uint32_t* __restrict__ ww =
        (const uint32_t*)wf_t +
        (((size_t)c * WSTRIDE + ((size_t)(Q0 + WCHPAD - j) << 4) + ((size_t)h << 3)) >> 1);

    v8f acc = {};

#pragma unroll 2
    for (int t = 0; t < NITER; ++t) {
        pack32 ap;                         // A: 8 aligned b32 loads, imm offsets
        ap.u[0] = xw[0];
        ap.u[1] = xw[1];
        ap.u[2] = xw[2];
        ap.u[3] = xw[3];
        ap.u[4] = xw[8];
        ap.u[5] = xw[9];
        ap.u[6] = xw[10];
        ap.u[7] = xw[11];

        pack32 bp;                         // B: 8-half runs at 8h, 16+8h, 2 aligned b128 loads
        const uint4* __restrict__ bq = (const uint4*)ww;
        bp.q[0] = bq[0];
        bp.q[1] = bq[2];

        acc = __builtin_amdgcn_wmma_f32_16x16x32_f16(
            /*neg_a=*/false, ap.h, /*neg_b=*/false, bp.h,
            /*c_mod=*/(short)0, acc, /*reuse_a=*/false, /*reuse_b=*/false);

        xw += 16;                          // +32 halves: next Q (+2 chunks)
        ww += 16;
    }

    // epilogue: out[t,c] = relu(D + b[c]) * ei(c);  D element v -> M = v + 8h, N = j
    const float ei = (c < C - NUM_INH) ? 1.0f : -1.0f;
    const float bv = bias[c];
    const int tbase = (((g << 4) + j) << 4) + (h << 3);
#pragma unroll
    for (int v = 0; v < 8; ++v) {
        float val = acc[v] + bv;
        val = val > 0.f ? val : 0.f;
        out[(size_t)(tbase + v) * C + c] = val * ei;
    }
}

// ---------------- launch ----------------
extern "C" void kernel_launch(void* const* d_in, const int* in_sizes, int n_in,
                              void* d_out, int out_size, void* d_ws, size_t ws_size,
                              hipStream_t stream) {
    (void)in_sizes; (void)n_in; (void)out_size; (void)ws_size;
    const float* x = (const float*)d_in[0];   // (T, C)
    const float* w = (const float*)d_in[1];   // (FW, C)
    const float* b = (const float*)d_in[2];   // (C,)
    float* out = (float*)d_out;               // (T, C)

    // workspace: x2 (2 parity copies, 2*CX halves = ~49 MB) then padded wf (~17.5 MB)
    _Float16* x2   = (_Float16*)d_ws;
    _Float16* wf_t = (_Float16*)((char*)d_ws + 2 * CX * sizeof(_Float16));

    k_norm_wf<<<C, 256, 0, stream>>>(w, wf_t);
    k_x2<<<(unsigned)((2 * CX + 255) / 256), 256, 0, stream>>>(x, x2);
    k_conv<<<(C * 16) / 8, 256, 0, stream>>>(x2, wf_t, b, out);
}
